// GCN_38371237822486
// MI455X (gfx1250) — compile-verified, matrix operand order checked by reading, not run on hardware
//
#include <hip/hip_runtime.h>
#include <math.h>

// ---------- types for CDNA5 WMMA ----------
typedef __attribute__((ext_vector_type(16))) __bf16 v16bf;
typedef __attribute__((ext_vector_type(8)))  __bf16 v8bf;
typedef __attribute__((ext_vector_type(8)))  float  v8f;

#define BN_EPS 1e-5f

__device__ __forceinline__ unsigned short f2bf(float f) {
    union { float f; unsigned u; } v; v.f = f;
    unsigned u = v.u;
    unsigned r = (u + 0x7FFFu + ((u >> 16) & 1u)) >> 16;      // round-nearest-even
    if ((u & 0x7F800000u) == 0x7F800000u) r = u >> 16;        // inf/nan passthrough
    return (unsigned short)r;
}

// ---------- small utility kernels ----------
__global__ void k_fill_zero(float* __restrict__ p, long long n) {
    long long i = blockIdx.x * (long long)blockDim.x + threadIdx.x;
    if (i < n) p[i] = 0.0f;
}

__global__ void k_deg_init(float* __restrict__ deg, int n) {
    int i = blockIdx.x * blockDim.x + threadIdx.x;
    if (i < n) deg[i] = 1.0f;                                  // self loop
}

__global__ void k_deg_accum(const int* __restrict__ row, float* __restrict__ deg, int E) {
    int e = blockIdx.x * blockDim.x + threadIdx.x;
    if (e < E) atomicAdd(&deg[row[e]], 1.0f);
}

__global__ void k_dinv(float* __restrict__ d, int n) {
    int i = blockIdx.x * blockDim.x + threadIdx.x;
    if (i < n) { float v = d[i]; d[i] = (v > 0.0f) ? rsqrtf(v) : 0.0f; }
}

__global__ void k_norm(const int* __restrict__ row, const int* __restrict__ col,
                       const float* __restrict__ dinv, float* __restrict__ norm, int E) {
    int e = blockIdx.x * blockDim.x + threadIdx.x;
    if (e < E) norm[e] = dinv[row[e]] * dinv[col[e]];
}

// fp32 -> bf16 row-major copy with zero-padded tail rows
__global__ void k_f32_to_bf16(const float* __restrict__ src, unsigned short* __restrict__ dst,
                              int n, int npad, int C) {
    long long i = blockIdx.x * (long long)blockDim.x + threadIdx.x;
    long long total = (long long)npad * C;
    if (i >= total) return;
    int r = (int)(i / C);
    dst[i] = (r < n) ? f2bf(src[i]) : (unsigned short)0;
}

// Pack weight W[K x noutSrc] (row-major f32) into per-lane WMMA B-fragment layout:
// dst[((nt*ktiles + kt)*32 + lane)*16 + j]
__global__ void k_pack_w(const float* __restrict__ W, unsigned short* __restrict__ dst,
                         int K, int noutSrc, int ntiles) {
    int t = blockIdx.x * blockDim.x + threadIdx.x;
    int ktiles = K >> 5;
    int total = ntiles * ktiles * 32;
    if (t >= total) return;
    int lane = t & 31;
    int kt = (t >> 5) % ktiles;
    int nt = (t >> 5) / ktiles;
    int half = lane >> 4;
    int ncol = nt * 16 + (lane & 15);
    unsigned short* o = dst + (long long)t * 16;
#pragma unroll
    for (int j = 0; j < 8; ++j) {
        int k0 = kt * 32 + 8 * half + j;
        int k1 = kt * 32 + 16 + 8 * half + j;
        o[j]     = (ncol < noutSrc) ? f2bf(W[(long long)k0 * noutSrc + ncol]) : (unsigned short)0;
        o[8 + j] = (ncol < noutSrc) ? f2bf(W[(long long)k1 * noutSrc + ncol]) : (unsigned short)0;
    }
}

// ---------- WMMA GEMM: C[npad x Nout] = A[npad x K](bf16) * Bpacked ----------
// block = 256 threads (8 waves). Wave w handles rows [blockIdx.x*128 + 16w, +16),
// and 4 column tiles of 16 starting at blockIdx.y*64. K-loop in steps of 32.
__global__ void k_wmma_gemm(const unsigned short* __restrict__ Abf,
                            const unsigned short* __restrict__ Bp,
                            float* __restrict__ C, int K, int Nout) {
    const int wave = threadIdx.x >> 5;
    const int lane = threadIdx.x & 31;
    const int half = lane >> 4;
    const int mrow = lane & 15;
    const int mbase = blockIdx.x * 128 + wave * 16;
    const int nbase = blockIdx.y * 64;
    const int ktiles = K >> 5;

    const __bf16* A = (const __bf16*)Abf;
    const __bf16* B = (const __bf16*)Bp;

    v8f acc[4];
#pragma unroll
    for (int i = 0; i < 4; ++i) acc[i] = (v8f)0.0f;

    for (int kt = 0; kt < ktiles; ++kt) {
        // A fragment: lane holds row (mbase+mrow); elems 0-7 = K[kt*32+8h .. +7],
        // elems 8-15 = K[kt*32+16+8h .. +7]  (ISA 7.12.2, 16-bit A 16x32)
        const __bf16* ap = A + (long long)(mbase + mrow) * K + kt * 32 + 8 * half;
        union { v16bf v; v8bf h[2]; } a;
        a.h[0] = *(const v8bf*)(ap);
        a.h[1] = *(const v8bf*)(ap + 16);
#pragma unroll
        for (int nt = 0; nt < 4; ++nt) {
            int ntile = (nbase >> 4) + nt;
            v16bf b = *(const v16bf*)(B + ((long long)(ntile * ktiles + kt) * 32 + lane) * 16);
            acc[nt] = __builtin_amdgcn_wmma_f32_16x16x32_bf16(
                false, a.v, false, b, (short)0, acc[nt], false, false);
        }
    }
    // C/D layout: VGPR r -> row (8*half + r), col = lane&15 within tile
#pragma unroll
    for (int nt = 0; nt < 4; ++nt) {
#pragma unroll
        for (int r = 0; r < 8; ++r) {
            C[(long long)(mbase + 8 * half + r) * Nout + nbase + nt * 16 + (lane & 15)] = acc[nt][r];
        }
    }
}

// ---------- edge aggregation: agg[row] += w * h[col] (float4 chunks) ----------
// logical edges [0,E) are real edges (weight norm[e]); [E, E+n) are self loops (w = dinv^2)
__global__ void k_aggregate(const float* __restrict__ h,
                            const int* __restrict__ row, const int* __restrict__ col,
                            const float* __restrict__ norm, const float* __restrict__ dinv,
                            float* __restrict__ agg, int E, int n, int C, int ldh) {
    unsigned tid = blockIdx.x * blockDim.x + threadIdx.x;
    unsigned cpv = (unsigned)(C >> 2);
    unsigned total = (unsigned)(E + n) * cpv;
    if (tid >= total) return;
    unsigned e = tid / cpv;
    int c = (int)(tid - e * cpv) * 4;
    int r, s; float w;
    if ((int)e < E) { r = row[e]; s = col[e]; w = norm[e]; }
    else            { r = s = (int)e - E; float di = dinv[r]; w = di * di; }
    const float4 v = *(const float4*)(h + (long long)s * ldh + c);
    float* o = agg + (long long)r * C + c;
    atomicAdd(o + 0, w * v.x);
    atomicAdd(o + 1, w * v.y);
    atomicAdd(o + 2, w * v.z);
    atomicAdd(o + 3, w * v.w);
}

// ---------- batchnorm stats: per-channel sum & sumsq (C == blockDim.x) ----------
__global__ void k_bn_stats(const float* __restrict__ h, float* __restrict__ sums,
                           int n, int C, int rowsPerBlock) {
    int c = threadIdx.x;
    int r0 = blockIdx.x * rowsPerBlock;
    int r1 = min(n, r0 + rowsPerBlock);
    float s = 0.0f, s2 = 0.0f;
    for (int r = r0; r < r1; ++r) {
        float v = h[(long long)r * C + c];
        s += v; s2 += v * v;
    }
    atomicAdd(&sums[c], s);
    atomicAdd(&sums[C + c], s2);
}

// ---------- fused BN + ReLU + fp32->bf16 (zero-padded tail rows) ----------
__global__ void k_bn_relu_bf16(const float* __restrict__ agg, const float* __restrict__ sums,
                               const float* __restrict__ g, const float* __restrict__ be,
                               unsigned short* __restrict__ out,
                               int n, int npad, int C, float invN) {
    long long i = blockIdx.x * (long long)blockDim.x + threadIdx.x;
    long long total = (long long)npad * C;
    if (i >= total) return;
    int r = (int)(i / C);
    int c = (int)(i - (long long)r * C);
    if (r >= n) { out[i] = 0; return; }
    float mu  = sums[c] * invN;
    float var = sums[C + c] * invN - mu * mu;
    float y = (agg[i] - mu) * rsqrtf(var + BN_EPS) * g[c] + be[c];
    out[i] = f2bf(fmaxf(y, 0.0f));
}

// ---------- final bias + log_softmax over 40 classes, one wave per row ----------
__global__ void k_logsoftmax(const float* __restrict__ agg, const float* __restrict__ b3,
                             float* __restrict__ out, int n) {
    int wave = threadIdx.x >> 5;
    int lane = threadIdx.x & 31;
    int r = blockIdx.x * 4 + wave;
    if (r >= n) return;
    const float* in = agg + (long long)r * 40;
    float v0 = in[lane] + b3[lane];                          // lane < 32 < 40 always valid
    float v1 = (lane + 32 < 40) ? (in[lane + 32] + b3[lane + 32]) : -INFINITY;
    float m = fmaxf(v0, v1);
#pragma unroll
    for (int o = 16; o > 0; o >>= 1) m = fmaxf(m, __shfl_xor(m, o, 32));
    float s = __expf(v0 - m) + ((lane + 32 < 40) ? __expf(v1 - m) : 0.0f);
#pragma unroll
    for (int o = 16; o > 0; o >>= 1) s += __shfl_xor(s, o, 32);
    float ls = __logf(s);
    float* op = out + (long long)r * 40;
    op[lane] = v0 - m - ls;
    if (lane + 32 < 40) op[lane + 32] = v1 - m - ls;
}

// =====================================================================
extern "C" void kernel_launch(void* const* d_in, const int* in_sizes, int n_in,
                              void* d_out, int out_size, void* d_ws, size_t ws_size,
                              hipStream_t stream) {
    const float* x  = (const float*)d_in[0];
    const int*   ei = (const int*)d_in[1];
    const float* W1 = (const float*)d_in[2];
    const float* g1 = (const float*)d_in[4];
    const float* be1= (const float*)d_in[5];
    const float* W2 = (const float*)d_in[6];
    const float* g2 = (const float*)d_in[8];
    const float* be2= (const float*)d_in[9];
    const float* W3 = (const float*)d_in[10];
    const float* b3 = (const float*)d_in[11];

    const int n = in_sizes[0] / 128;
    const int E = in_sizes[1] / 2;
    const int npad = ((n + 127) / 128) * 128;
    const int* row = ei;        // destinations
    const int* col = ei + E;    // sources

    // ---- workspace carve-up ----
    char* cur = (char*)d_ws;
    auto alloc = [&](size_t bytes) -> void* {
        void* p = (void*)cur;
        cur += (bytes + 255) & ~(size_t)255;
        return p;
    };
    float*          dinv  = (float*)alloc((size_t)npad * 4);
    float*          norm  = (float*)alloc((size_t)E * 4);
    float*          bufA  = (float*)alloc((size_t)npad * 256 * 4);   // GEMM out
    float*          bufB  = (float*)alloc((size_t)npad * 256 * 4);   // aggregation out
    unsigned short* bufC  = (unsigned short*)alloc((size_t)npad * 256 * 2); // bf16 acts
    unsigned short* W1p   = (unsigned short*)alloc((size_t)16 * 4 * 32 * 16 * 2);
    unsigned short* W2p   = (unsigned short*)alloc((size_t)16 * 8 * 32 * 16 * 2);
    unsigned short* W3p   = (unsigned short*)alloc((size_t)4  * 8 * 32 * 16 * 2);
    float*          stats = (float*)alloc(512 * 4);

    const int TB = 256;
    auto blocks = [](long long work, int tb) { return (unsigned)((work + tb - 1) / tb); };

    // ---- graph normalization ----
    k_deg_init <<<blocks(n, TB), TB, 0, stream>>>(dinv, n);
    k_deg_accum<<<blocks(E, TB), TB, 0, stream>>>(row, dinv, E);
    k_dinv     <<<blocks(n, TB), TB, 0, stream>>>(dinv, n);
    k_norm     <<<blocks(E, TB), TB, 0, stream>>>(row, col, dinv, norm, E);

    // ---- operand prep ----
    k_f32_to_bf16<<<blocks((long long)npad * 128, TB), TB, 0, stream>>>(x, bufC, n, npad, 128);
    k_pack_w<<<blocks(16 * 4 * 32, TB), TB, 0, stream>>>(W1, W1p, 128, 256, 16);
    k_pack_w<<<blocks(16 * 8 * 32, TB), TB, 0, stream>>>(W2, W2p, 256, 256, 16);
    k_pack_w<<<blocks(4  * 8 * 32, TB), TB, 0, stream>>>(W3, W3p, 256, 40, 4); // pad 40->64

    const float invN = 1.0f / (float)n;
    const int gemmGX = npad / 128;

    // ================= layer 1: 128 -> 256 =================
    k_wmma_gemm<<<dim3(gemmGX, 4), TB, 0, stream>>>(bufC, W1p, bufA, 128, 256);
    k_fill_zero<<<blocks((long long)npad * 256, TB), TB, 0, stream>>>(bufB, (long long)npad * 256);
    k_aggregate<<<blocks((long long)(E + n) * 64, TB), TB, 0, stream>>>(
        bufA, row, col, norm, dinv, bufB, E, n, 256, 256);
    k_fill_zero<<<2, TB, 0, stream>>>(stats, 512);
    k_bn_stats<<<blocks(n, 128), 256, 0, stream>>>(bufB, stats, n, 256, 128);
    k_bn_relu_bf16<<<blocks((long long)npad * 256, TB), TB, 0, stream>>>(
        bufB, stats, g1, be1, bufC, n, npad, 256, invN);

    // ================= layer 2: 256 -> 256 =================
    k_wmma_gemm<<<dim3(gemmGX, 4), TB, 0, stream>>>(bufC, W2p, bufA, 256, 256);
    k_fill_zero<<<blocks((long long)npad * 256, TB), TB, 0, stream>>>(bufB, (long long)npad * 256);
    k_aggregate<<<blocks((long long)(E + n) * 64, TB), TB, 0, stream>>>(
        bufA, row, col, norm, dinv, bufB, E, n, 256, 256);
    k_fill_zero<<<2, TB, 0, stream>>>(stats, 512);
    k_bn_stats<<<blocks(n, 128), 256, 0, stream>>>(bufB, stats, n, 256, 128);
    k_bn_relu_bf16<<<blocks((long long)npad * 256, TB), TB, 0, stream>>>(
        bufB, stats, g2, be2, bufC, n, npad, 256, invN);

    // ================= layer 3: 256 -> 40 (padded to 64) =================
    k_wmma_gemm<<<dim3(gemmGX, 1), TB, 0, stream>>>(bufC, W3p, bufA, 256, 64);
    k_fill_zero<<<blocks((long long)n * 40, TB), TB, 0, stream>>>(bufB, (long long)n * 40);
    k_aggregate<<<blocks((long long)(E + n) * 10, TB), TB, 0, stream>>>(
        bufA, row, col, norm, dinv, bufB, E, n, 40, 64);

    // ---- bias + log_softmax -> d_out ----
    k_logsoftmax<<<blocks(n, 4) , 128, 0, stream>>>(bufB, b3, (float*)d_out, n);
}
